// MemoryOptimizedMLA_27822798144109
// MI455X (gfx1250) — matrix operand order repaired and checked by reading, not compiled, not run on hardware
//
#include <hip/hip_runtime.h>

// ---------------------------------------------------------------------------
// MLA forward for MI455X (gfx1250).
// All GEMMs via v_wmma_f32_16x16x32_bf16 (f32 accumulate), double-buffered
// LDS tiles staged with GLOBAL_LOAD_ASYNC_TO_LDS_B128 (ASYNCcnt-tracked),
// bank-conflict-padded fragment layout, distance-2 pipelined B-fragments.
// fp32 inputs cast to bf16 once; weights transposed at cast time so both
// GEMM operands stage as contiguous 16-byte row segments.
// ---------------------------------------------------------------------------

typedef __bf16 bf16;
typedef __attribute__((ext_vector_type(16))) __bf16        v16bf;
typedef __attribute__((ext_vector_type(8)))  float         v8f;
typedef __attribute__((ext_vector_type(8)))  unsigned int  v8u;
typedef int v4i_vs __attribute__((vector_size(16)));        // matches builtin param

union ABFrag { v8u u; v16bf b; };

constexpr int D_MODEL_C = 5120;
constexpr int N_HEADS_C = 8;
constexpr int D_HEAD_C  = 640;
constexpr int D_ROPE_C  = 16;
constexpr int SPLIT_C   = 624;   // D_HEAD - D_ROPE
constexpr int D_KV_C    = 128;
constexpr int BATCH_C   = 2;
constexpr int SEQ_C     = 2048;
constexpr int MROWS_C   = BATCH_C * SEQ_C;   // 4096

#define AS_GLOBAL __attribute__((address_space(1)))
#define AS_LDS    __attribute__((address_space(3)))

#if defined(__HIP_DEVICE_COMPILE__) && __has_builtin(__builtin_amdgcn_global_load_async_to_lds_b128)
#define USE_ASYNC_LDS 1
#else
#define USE_ASYNC_LDS 0
#endif

#if USE_ASYNC_LDS
#if __has_builtin(__builtin_amdgcn_s_wait_asynccnt)
#define WAIT_ASYNC(n) __builtin_amdgcn_s_wait_asynccnt(n)
#else
#define WAIT_ASYNC(n) asm volatile("s_wait_asynccnt %0" :: "n"(n) : "memory")
#endif
#define ASYNC_CP16(gptr, lptr) \
  __builtin_amdgcn_global_load_async_to_lds_b128( \
      (AS_GLOBAL v4i_vs*)(void*)(gptr), (AS_LDS v4i_vs*)(void*)(lptr), 0, 0)
#endif

// --------------------------- fp32 -> bf16 cast -----------------------------
__global__ void cast_f32_to_bf16(const float* __restrict__ src,
                                 bf16* __restrict__ dst, size_t n) {
  size_t i  = (size_t)blockIdx.x * blockDim.x + threadIdx.x;
  size_t st = (size_t)gridDim.x * blockDim.x;
  for (; i < n; i += st) dst[i] = (bf16)src[i];
}

// ---------------- fp32 [R][C] -> bf16 [C][R] tiled transpose ---------------
__global__ __launch_bounds__(256)
void transpose_cast_f32_bf16(const float* __restrict__ in,
                             bf16* __restrict__ outp, int R, int C) {
  __shared__ float tile[32][33];
  const int c0 = blockIdx.x * 32, r0 = blockIdx.y * 32;
  const int tx = threadIdx.x, ty = threadIdx.y;   // 32 x 8
  #pragma unroll
  for (int i = 0; i < 32; i += 8)
    tile[ty + i][tx] = in[(size_t)(r0 + ty + i) * C + c0 + tx];
  __syncthreads();
  #pragma unroll
  for (int i = 0; i < 32; i += 8)
    outp[(size_t)(c0 + ty + i) * R + r0 + tx] = (bf16)tile[tx][ty + i];
}

// ------------- v [B,S,H*640] -> vT [B,H,640,SEQ] tiled transpose -----------
__global__ __launch_bounds__(256)
void transpose_v(const bf16* __restrict__ in, bf16* __restrict__ outp) {
  __shared__ bf16 tile[32][33];
  const int z = blockIdx.z, b = z >> 3, h = z & 7;
  const bf16* src = in + (size_t)b * SEQ_C * D_MODEL_C + h * D_HEAD_C;
  bf16* dst = outp + (size_t)z * D_HEAD_C * SEQ_C;
  const int d0 = blockIdx.x * 32, s0 = blockIdx.y * 32;
  const int tx = threadIdx.x, ty = threadIdx.y;   // 32 x 8
  #pragma unroll
  for (int i = 0; i < 32; i += 8)
    tile[ty + i][tx] = src[(size_t)(s0 + ty + i) * D_MODEL_C + d0 + tx];
  __syncthreads();
  #pragma unroll
  for (int i = 0; i < 32; i += 8)
    dst[(size_t)(d0 + ty + i) * SEQ_C + s0 + tx] = tile[tx][ty + i];
}

// ------------------------------ WMMA GEMM ----------------------------------
// C[M,N] = scale * (A[M,K] @ B[N,K]^T) + bias   (B always stored [N][K]).
// Batched over blockIdx.z: off = (z/nh)*sXb + (z%nh)*sXh.
// 128x128 block tile, 8 waves, each wave owns a 16x128 strip.
// Double-buffered LDS, row stride padded to 20 dwords (16 distinct banks,
// keeps 16B alignment for ds_load_b128 fragment reads).
template<bool OUT_BF16>
__global__ __launch_bounds__(256)
void wmma_gemm(const bf16* __restrict__ Ag, long long lda, long long sAb, long long sAh,
               const bf16* __restrict__ Bg, long long ldb, long long sBb, long long sBh,
               void* __restrict__ Cg, long long ldc, long long sCb, long long sCh,
               const float* __restrict__ bias, float scale, int K, int nh)
{
  constexpr int LDT  = 20;          // padded u32 row stride
  constexpr int BUFW = 128 * LDT;   // u32 per buffer
  __shared__ unsigned int a32[2 * BUFW];
  __shared__ unsigned int b32[2 * BUFW];

  const int z  = blockIdx.z;
  const int zb = z / nh, zh = z % nh;
  const bf16* A = Ag + (size_t)zb * sAb + (size_t)zh * sAh;
  const bf16* B = Bg + (size_t)zb * sBb + (size_t)zh * sBh;
  const size_t coff = (size_t)zb * sCb + (size_t)zh * sCh;

  const int tid  = threadIdx.x;
  const int wv   = tid >> 5;        // wave id (wave32)
  const int lane = tid & 31;
  const int hlf  = lane >> 4;
  const int l15  = lane & 15;
  const int Mb   = blockIdx.y * 128;
  const int Nb   = blockIdx.x * 128;

  v8f c[8];
  #pragma unroll
  for (int j = 0; j < 8; ++j)
    #pragma unroll
    for (int e = 0; e < 8; ++e) c[j][e] = 0.0f;

  // staging: 2 threads per row, 16 bf16 (2 x 16B) each
  const int am  = tid >> 1;
  const int akh = tid & 1;
  const bf16* Arow = A + (size_t)(Mb + am) * lda + akh * 16;
  const bf16* Brow = B + (size_t)(Nb + am) * ldb + akh * 16;
  unsigned int* aL = &a32[am * LDT + akh * 8];
  unsigned int* bL = &b32[am * LDT + akh * 8];

  const int nt = K >> 5;            // K-steps of 32

  auto compute = [&](int cur) {
    const unsigned int* ab = &a32[cur * BUFW];
    const unsigned int* bb = &b32[cur * BUFW];
    ABFrag a;
    const int arow = wv * 16 + l15;
    #pragma unroll
    for (int r = 0; r < 8; ++r) {
      const int pidx = hlf * 8 + r;
      a.u[r] = ab[arow * LDT + pidx];
    }
    // B-fragments: 3-slot rotation, prefetch distance 2, so each WMMA's
    // operand was loaded two WMMAs earlier (4 younger ds loads may remain
    // outstanding at the wait point instead of a full dscnt drain).
    ABFrag bfr[3];
    #pragma unroll
    for (int r = 0; r < 8; ++r) bfr[0].u[r] = bb[(0 * 16 + l15) * LDT + hlf * 8 + r];
    #pragma unroll
    for (int r = 0; r < 8; ++r) bfr[1].u[r] = bb[(1 * 16 + l15) * LDT + hlf * 8 + r];
    #pragma unroll
    for (int j = 0; j < 8; ++j) {
      if (j + 2 < 8) {
        const int bcol = (j + 2) * 16 + l15;
        #pragma unroll
        for (int r = 0; r < 8; ++r) bfr[(j + 2) % 3].u[r] = bb[bcol * LDT + hlf * 8 + r];
      }
      c[j] = __builtin_amdgcn_wmma_f32_16x16x32_bf16(false, a.b, false, bfr[j % 3].b,
                                                     (short)0, c[j], false, false);
    }
  };

#if USE_ASYNC_LDS
  auto issue = [&](int kt, int buf) {       // 4 async 16B copies per thread
    const bf16* ga = Arow + kt;
    const bf16* gb = Brow + kt;
    unsigned int* la = aL + buf * BUFW;
    unsigned int* lb = bL + buf * BUFW;
    ASYNC_CP16(ga,     la);
    ASYNC_CP16(ga + 8, la + 4);
    ASYNC_CP16(gb,     lb);
    ASYNC_CP16(gb + 8, lb + 4);
  };

  issue(0, 0);
  for (int t = 0; t < nt; ++t) {
    const int cur = t & 1;
    if (t + 1 < nt) { issue((t + 1) << 5, cur ^ 1); WAIT_ASYNC(4); }
    else            { WAIT_ASYNC(0); }
    __syncthreads();                 // tile t visible to all waves
    compute(cur);
    __syncthreads();                 // readers done before buf reuse
  }
#else
  uint4 ra0, ra1, rb0, rb1;
  auto loadregs = [&](int kt) {
    const uint4* sa = (const uint4*)(Arow + kt);
    ra0 = sa[0]; ra1 = sa[1];
    const uint4* sb = (const uint4*)(Brow + kt);
    rb0 = sb[0]; rb1 = sb[1];
  };
  auto storeregs = [&](int buf) {
    uint4* la = (uint4*)(aL + buf * BUFW);
    la[0] = ra0; la[1] = ra1;
    uint4* lb = (uint4*)(bL + buf * BUFW);
    lb[0] = rb0; lb[1] = rb1;
  };

  loadregs(0);
  storeregs(0);
  __syncthreads();
  for (int t = 0; t < nt; ++t) {
    const int cur = t & 1;
    const bool nxt = (t + 1 < nt);
    if (nxt) loadregs((t + 1) << 5);  // overlap global loads with compute
    compute(cur);
    __syncthreads();
    if (nxt) storeregs(cur ^ 1);
    __syncthreads();
  }
#endif

  // epilogue: scale + bias, write f32 or bf16
  float* Cf = (float*)Cg;
  bf16*  Cb = (bf16*)Cg;
  #pragma unroll
  for (int j = 0; j < 8; ++j) {
    const int colg = Nb + j * 16 + l15;
    const float bv = bias ? bias[colg] : 0.0f;
    #pragma unroll
    for (int r = 0; r < 8; ++r) {
      const int rowg = Mb + wv * 16 + r + hlf * 8;
      const float val = c[j][r] * scale + bv;
      const size_t idx = coff + (size_t)rowg * ldc + colg;
      if (OUT_BF16) Cb[idx] = (bf16)val; else Cf[idx] = val;
    }
  }
}

// ------------------------------ softmax ------------------------------------
__global__ __launch_bounds__(256)
void softmax_rows(const float* __restrict__ scores, bf16* __restrict__ attn) {
  __shared__ float red[256];
  const size_t row = blockIdx.x;
  const float* src = scores + row * (size_t)SEQ_C;
  bf16* dst = attn + row * (size_t)SEQ_C;
  const int tid = threadIdx.x;

  float v[8];
  float m = -3.0e38f;
  #pragma unroll
  for (int i = 0; i < 8; ++i) { v[i] = src[tid + 256 * i]; m = fmaxf(m, v[i]); }
  red[tid] = m; __syncthreads();
  for (int s = 128; s > 0; s >>= 1) {
    if (tid < s) red[tid] = fmaxf(red[tid], red[tid + s]);
    __syncthreads();
  }
  m = red[0]; __syncthreads();

  float sum = 0.0f;
  #pragma unroll
  for (int i = 0; i < 8; ++i) { v[i] = __expf(v[i] - m); sum += v[i]; }
  red[tid] = sum; __syncthreads();
  for (int s = 128; s > 0; s >>= 1) {
    if (tid < s) red[tid] += red[tid + s];
    __syncthreads();
  }
  const float inv = 1.0f / red[0];
  #pragma unroll
  for (int i = 0; i < 8; ++i) dst[tid + 256 * i] = (bf16)(v[i] * inv);
}

// --------------------------- rope + assemble -------------------------------
__global__ void rope_assemble(const bf16* __restrict__ base,
                              const float* __restrict__ rot,
                              bf16* __restrict__ outq, size_t total) {
  size_t idx = (size_t)blockIdx.x * blockDim.x + threadIdx.x;
  if (idx >= total) return;
  const int d = (int)(idx % D_HEAD_C);
  const size_t t2 = idx / D_HEAD_C;
  const int h = (int)(t2 % N_HEADS_C);
  const size_t row = t2 / N_HEADS_C;       // b*SEQ + s
  const int s = (int)(row % SEQ_C);

  bf16 o;
  if (d < SPLIT_C) {
    o = base[row * (size_t)(N_HEADS_C * SPLIT_C) + h * SPLIT_C + d];
  } else {
    const int dr = d - SPLIT_C;
    const float* x = rot + row * (size_t)(N_HEADS_C * D_ROPE_C) + h * D_ROPE_C;
    float val;
    if (dr < 8) {
      const int j = dr & 3;                                        // 4 freqs
      const float invf = __expf(-9.210340372f * 0.25f * (float)j); // 1e4^(-j/4)
      const float ang = ((float)s / 40.0f) * invf;
      const float cs = __cosf(ang), sn = __sinf(ang);
      const float rh = (dr < 4) ? -x[dr + 4] : x[dr - 4];          // rotate_half
      val = x[dr] * cs + rh * sn;
    } else {
      val = x[dr];
    }
    o = (bf16)val;
  }
  outq[idx] = o;
}

// ------------------------------ launchers ----------------------------------
template<bool OUT_BF16>
static void run_gemm(hipStream_t stream, int Mm, int Nn, int K, int batch, int nh,
                     const bf16* A, long long lda, long long sAb, long long sAh,
                     const bf16* B, long long ldb, long long sBb, long long sBh,
                     void* C, long long ldc, long long sCb, long long sCh,
                     const float* bias, float scale)
{
  dim3 grid(Nn / 128, Mm / 128, batch);
  wmma_gemm<OUT_BF16><<<grid, 256, 0, stream>>>(
      A, lda, sAb, sAh, B, ldb, sBb, sBh, C, ldc, sCb, sCh, bias, scale, K, nh);
}

extern "C" void kernel_launch(void* const* d_in, const int* in_sizes, int n_in,
                              void* d_out, int out_size, void* d_ws, size_t ws_size,
                              hipStream_t stream)
{
  (void)in_sizes; (void)n_in; (void)out_size; (void)ws_size;

  const float* h_in = (const float*)d_in[0];
  const float* Wdkv = (const float*)d_in[1];
  const float* bdkv = (const float*)d_in[2];
  const float* Wdq  = (const float*)d_in[3];
  const float* bdq  = (const float*)d_in[4];
  const float* Wuk  = (const float*)d_in[5];
  const float* buk  = (const float*)d_in[6];
  const float* Wuv  = (const float*)d_in[7];
  const float* buv  = (const float*)d_in[8];
  const float* Wuq  = (const float*)d_in[9];
  const float* buq  = (const float*)d_in[10];
  const float* Wqr  = (const float*)d_in[11];
  const float* bqr  = (const float*)d_in[12];
  const float* Wkr  = (const float*)d_in[13];
  const float* bkr  = (const float*)d_in[14];
  const float* Wo   = (const float*)d_in[15];
  const float* bo   = (const float*)d_in[16];
  float* out = (float*)d_out;

  // ---- carve workspace ----
  char* ws = (char*)d_ws;
  size_t off = 0;
  auto alloc = [&](size_t bytes) -> void* {
    void* p = ws + off;
    off += (bytes + 255) & ~(size_t)255;
    return p;
  };
  const size_t M = MROWS_C;
  // transposed bf16 weights (WT[n][k])
  bf16* h_bf     = (bf16*)alloc(M * D_MODEL_C * 2);
  bf16* wdkvT    = (bf16*)alloc((size_t)D_KV_C * D_MODEL_C * 2);
  bf16* wdqT     = (bf16*)alloc((size_t)D_KV_C * D_MODEL_C * 2);
  bf16* wukT     = (bf16*)alloc((size_t)(N_HEADS_C * SPLIT_C) * D_KV_C * 2);
  bf16* wuvT     = (bf16*)alloc((size_t)D_MODEL_C * D_KV_C * 2);
  bf16* wuqT     = (bf16*)alloc((size_t)(N_HEADS_C * SPLIT_C) * D_KV_C * 2);
  bf16* wqrT     = (bf16*)alloc((size_t)D_KV_C * D_KV_C * 2);
  bf16* wkrT     = (bf16*)alloc((size_t)D_KV_C * D_MODEL_C * 2);
  bf16* woT      = (bf16*)alloc((size_t)D_MODEL_C * D_MODEL_C * 2);
  bf16* ckv_bf   = (bf16*)alloc(M * D_KV_C * 2);
  bf16* cq_bf    = (bf16*)alloc(M * D_KV_C * 2);
  bf16* kbase_bf = (bf16*)alloc(M * (N_HEADS_C * SPLIT_C) * 2);
  bf16* qbase_bf = (bf16*)alloc(M * (N_HEADS_C * SPLIT_C) * 2);
  bf16* v_bf     = (bf16*)alloc(M * D_MODEL_C * 2);
  bf16* vT_bf    = (bf16*)alloc(M * D_MODEL_C * 2);   // [B,H,640,SEQ]
  float* qrot_f  = (float*)alloc(M * D_KV_C * 4);
  float* krot_f  = (float*)alloc(M * D_KV_C * 4);
  bf16* q_bf     = (bf16*)alloc(M * D_MODEL_C * 2);
  bf16* k_bf     = (bf16*)alloc(M * D_MODEL_C * 2);
  float* scores  = (float*)alloc((size_t)BATCH_C * N_HEADS_C * SEQ_C * SEQ_C * 4);
  bf16* attn_bf  = (bf16*)alloc((size_t)BATCH_C * N_HEADS_C * SEQ_C * SEQ_C * 2);
  bf16* ctx_bf   = (bf16*)alloc(M * D_MODEL_C * 2);

  auto tcast = [&](const float* src, bf16* dst, int R, int C) {  // [R][C]->[C][R]
    dim3 grid(C / 32, R / 32);
    transpose_cast_f32_bf16<<<grid, dim3(32, 8), 0, stream>>>(src, dst, R, C);
  };

  // ---- 0) casts (h plain; weights transposed to [N][K]) ----
  {
    size_t n = M * D_MODEL_C;
    cast_f32_to_bf16<<<(int)((n + 1023) / 1024), 256, 0, stream>>>(h_in, h_bf, n);
  }
  tcast(Wdkv, wdkvT, D_MODEL_C, D_KV_C);
  tcast(Wdq,  wdqT,  D_MODEL_C, D_KV_C);
  tcast(Wuk,  wukT,  D_KV_C, N_HEADS_C * SPLIT_C);
  tcast(Wuv,  wuvT,  D_KV_C, D_MODEL_C);
  tcast(Wuq,  wuqT,  D_KV_C, N_HEADS_C * SPLIT_C);
  tcast(Wqr,  wqrT,  D_KV_C, D_KV_C);
  tcast(Wkr,  wkrT,  D_MODEL_C, D_KV_C);
  tcast(Wo,   woT,   D_MODEL_C, D_MODEL_C);

  // ---- 1) latent projections ----
  run_gemm<true>(stream, MROWS_C, D_KV_C, D_MODEL_C, 1, 1,
                 h_bf, D_MODEL_C, 0, 0, wdkvT, D_MODEL_C, 0, 0,
                 ckv_bf, D_KV_C, 0, 0, bdkv, 1.0f);
  run_gemm<true>(stream, MROWS_C, D_KV_C, D_MODEL_C, 1, 1,
                 h_bf, D_MODEL_C, 0, 0, wdqT, D_MODEL_C, 0, 0,
                 cq_bf, D_KV_C, 0, 0, bdq, 1.0f);

  // ---- 2) up projections ----
  run_gemm<true>(stream, MROWS_C, N_HEADS_C * SPLIT_C, D_KV_C, 1, 1,
                 ckv_bf, D_KV_C, 0, 0, wukT, D_KV_C, 0, 0,
                 kbase_bf, N_HEADS_C * SPLIT_C, 0, 0, buk, 1.0f);
  run_gemm<true>(stream, MROWS_C, D_MODEL_C, D_KV_C, 1, 1,
                 ckv_bf, D_KV_C, 0, 0, wuvT, D_KV_C, 0, 0,
                 v_bf, D_MODEL_C, 0, 0, buv, 1.0f);
  run_gemm<true>(stream, MROWS_C, N_HEADS_C * SPLIT_C, D_KV_C, 1, 1,
                 cq_bf, D_KV_C, 0, 0, wuqT, D_KV_C, 0, 0,
                 qbase_bf, N_HEADS_C * SPLIT_C, 0, 0, buq, 1.0f);
  run_gemm<false>(stream, MROWS_C, D_KV_C, D_KV_C, 1, 1,
                  cq_bf, D_KV_C, 0, 0, wqrT, D_KV_C, 0, 0,
                  qrot_f, D_KV_C, 0, 0, bqr, 1.0f);
  run_gemm<false>(stream, MROWS_C, D_KV_C, D_MODEL_C, 1, 1,
                  h_bf, D_MODEL_C, 0, 0, wkrT, D_MODEL_C, 0, 0,
                  krot_f, D_KV_C, 0, 0, bkr, 1.0f);

  // ---- 3) rope + concat into q / k ----
  const size_t qk_total = M * (size_t)(N_HEADS_C * D_HEAD_C);
  const int rope_blocks = (int)((qk_total + 255) / 256);
  rope_assemble<<<rope_blocks, 256, 0, stream>>>(qbase_bf, qrot_f, q_bf, qk_total);
  rope_assemble<<<rope_blocks, 256, 0, stream>>>(kbase_bf, krot_f, k_bf, qk_total);

  // ---- 3b) v -> vT per (b,h) for the PV GEMM ----
  transpose_v<<<dim3(D_HEAD_C / 32, SEQ_C / 32, BATCH_C * N_HEADS_C),
                dim3(32, 8), 0, stream>>>(v_bf, vT_bf);

  // ---- 4) scores = (q @ k^T) / sqrt(640), batched over (b,h) ----
  const long long sBSd = (long long)SEQ_C * D_MODEL_C;
  const long long sSS  = (long long)SEQ_C * SEQ_C;
  run_gemm<false>(stream, SEQ_C, SEQ_C, D_HEAD_C, BATCH_C * N_HEADS_C, N_HEADS_C,
                  q_bf, D_MODEL_C, sBSd, D_HEAD_C,
                  k_bf, D_MODEL_C, sBSd, D_HEAD_C,
                  scores, SEQ_C, (long long)N_HEADS_C * sSS, sSS,
                  nullptr, 0.03952847075f /* 1/sqrt(640) */);

  // ---- 5) softmax rows -> bf16 attn ----
  softmax_rows<<<BATCH_C * N_HEADS_C * SEQ_C, 256, 0, stream>>>(scores, attn_bf);

  // ---- 6) ctx = attn @ v  (B = vT[b,h,d,s], contract over s) ----
  run_gemm<true>(stream, SEQ_C, D_HEAD_C, SEQ_C, BATCH_C * N_HEADS_C, N_HEADS_C,
                 attn_bf, SEQ_C, (long long)N_HEADS_C * sSS, sSS,
                 vT_bf, SEQ_C, (long long)N_HEADS_C * D_HEAD_C * SEQ_C,
                 (long long)D_HEAD_C * SEQ_C,
                 ctx_bf, D_MODEL_C, sBSd, D_HEAD_C,
                 nullptr, 1.0f);

  // ---- 7) output projection (f32 out) ----
  run_gemm<false>(stream, MROWS_C, D_MODEL_C, D_MODEL_C, 1, 1,
                  ctx_bf, D_MODEL_C, 0, 0, woT, D_MODEL_C, 0, 0,
                  out, D_MODEL_C, 0, 0, bo, 1.0f);
}
